// WindowAttention_90022514524840
// MI455X (gfx1250) — matrix operand order repaired and checked by reading, not compiled, not run on hardware
//
#include <hip/hip_runtime.h>
#include <hip/hip_bf16.h>

typedef _Float16 v16h __attribute__((ext_vector_type(16)));
typedef _Float16 v8h  __attribute__((ext_vector_type(8)));
typedef _Float16 v4h  __attribute__((ext_vector_type(4)));
typedef float    v8f  __attribute__((ext_vector_type(8)));

// ---------------- constants ----------------
#define BATCH 8
#define IMG   224
#define C     96
#define WSZ   7
#define NWIN  8192            // 8 * 32 * 32
#define NH    3
#define HD    32
#define L     49
#define LP    64              // padded token count
#define XP    104             // pitch (halves) for [64 x 96] buffers, 16B aligned rows
#define PP    72              // pitch for P [64 x 64] and Vt [96 x 64]
#define QKV_FRAGS 54          // 18 ntiles * 3 ksteps
#define PRJ_FRAGS 18          // 6 ntiles * 3 ksteps
#define FRAG_HALVES 512       // 32 lanes * 16 halves

// -------- A-operand fragment: 16x32 f16, row-major source in LDS --------
__device__ inline v16h load_a_frag(const _Float16* buf, int pitch, int row,
                                   int k0, int kselA) {
    union { v16h v; v8h h[2]; } u;
    const _Float16* p = buf + row * pitch + k0 + 2 * kselA;   // kselA in {0,8} -> K offset {0,16}
    u.h[0] = *(const v8h*)(p);        // K = k0+2*ksel .. +7
    u.h[1] = *(const v8h*)(p + 8);    // K = k0+2*ksel+8 .. +15
    return u.v;
}

// -------- B-operand fragment: 16 contiguous halves per lane --------
__device__ inline v16h load_b_frag(const _Float16* p) {
    union { v16h v; v8h h[2]; } u;
    u.h[0] = *(const v8h*)(p);
    u.h[1] = *(const v8h*)(p + 8);
    return u.v;
}

// ============ weight pre-pack: fp32 -> f16 WMMA B-fragment layout ============
__global__ __launch_bounds__(32)
void pack_weights_kernel(const float* __restrict__ qkv_w,
                         const float* __restrict__ proj_w,
                         _Float16* __restrict__ wfrag) {
    const int b = blockIdx.x;        // 0..71
    const int l = threadIdx.x;       // lane
    const int cl = l & 15;
    const int kselB = (l < 16) ? 0 : 16;
    if (b < QKV_FRAGS) {
        const int nt = b / 3, ks = b % 3;
        const int col = nt * 16 + cl;
        const int k0  = ks * 32 + kselB;
        _Float16* dst = wfrag + (size_t)b * FRAG_HALVES + l * 16;
        #pragma unroll
        for (int j = 0; j < 16; ++j)
            dst[j] = (_Float16)qkv_w[(size_t)(k0 + j) * (3 * C) + col];
    } else {
        const int p = b - QKV_FRAGS;
        const int nt = p / 3, ks = p % 3;
        const int col = nt * 16 + cl;
        const int k0  = ks * 32 + kselB;
        _Float16* dst = wfrag + (size_t)b * FRAG_HALVES + l * 16;
        #pragma unroll
        for (int j = 0; j < 16; ++j)
            dst[j] = (_Float16)proj_w[(size_t)(k0 + j) * C + col];
    }
}

// ============ fused window attention: 1 block = 1 window, 3 waves ============
__global__ __launch_bounds__(96)
void win_attn_kernel(const float* __restrict__ x,
                     const _Float16* __restrict__ wfrag,
                     const float* __restrict__ qkv_b,
                     const float* __restrict__ proj_b,
                     float* __restrict__ out) {
    __shared__ alignas(16) _Float16 sX [LP * XP];       // X window f16; reused as O
    __shared__ alignas(16) _Float16 sQ [LP * XP];
    __shared__ alignas(16) _Float16 sK [LP * XP];
    __shared__ alignas(16) _Float16 sVt[C  * PP];       // V transposed: [head*32+hd][token]
    __shared__ alignas(16) _Float16 sP [NH * LP * PP];  // softmax probs per head
    _Float16* sO = sX;

    const int wid = blockIdx.x;
    const int bb  = wid >> 10;
    const int rem = wid & 1023;
    const int oy  = (rem >> 5) * WSZ;
    const int ox  = (rem & 31) * WSZ;

    const int tid  = threadIdx.x;
    const int lane = tid & 31;
    const int wv   = tid >> 5;                 // wave id 0..2
    const int cl   = lane & 15;                // N within tile (C/D), col lane
    const int rsel = (lane < 16) ? 0 : 8;      // C/D row offset
    const int ksA  = (lane < 16) ? 0 : 8;      // A operand K select
    const int ksB  = (lane < 16) ? 0 : 16;     // B operand K select

    const size_t xbase = (size_t)bb * IMG * IMG * C;

    // ---- stage 1: load window -> LDS f16, zero pad rows 49..63 ----
    for (int i = tid; i < L * (C / 4); i += 96) {
        const int t  = i / (C / 4);
        const int cc = (i % (C / 4)) * 4;
        const int gy = oy + t / WSZ, gx = ox + t % WSZ;
        const float4 v = *(const float4*)(x + xbase + ((size_t)gy * IMG + gx) * C + cc);
        v4h h = { (_Float16)v.x, (_Float16)v.y, (_Float16)v.z, (_Float16)v.w };
        *(v4h*)&sX[t * XP + cc] = h;
    }
    for (int i = tid; i < (LP - L) * (C / 4); i += 96) {
        const int t  = L + i / (C / 4);
        const int cc = (i % (C / 4)) * 4;
        v4h z = {};
        *(v4h*)&sX[t * XP + cc] = z;
    }
    __syncthreads();

    // ---- stage 2: QKV. wave 0 -> Q, wave 1 -> K, wave 2 -> V(transposed) ----
    // Branchless scatter: dst = base + row*rowMul + colOff (all wave-uniform selects,
    // no EXEC-mask juggling around the ds_store_b16s).
    {
        _Float16* dstBase = (wv == 0) ? sQ : ((wv == 1) ? sK : sVt);
        const int rowMul  = (wv == 2) ? 1 : XP;
        for (int mt = 0; mt < 4; ++mt) {
            for (int nt = 0; nt < 6; ++nt) {
                const int gnt = wv * 6 + nt;         // global N-tile in [0,18)
                v8f acc = {};
                #pragma unroll
                for (int ks = 0; ks < 3; ++ks) {
                    v16h a = load_a_frag(sX, XP, mt * 16 + cl, ks * 32, ksA);
                    v16h b = load_b_frag(wfrag + (size_t)(gnt * 3 + ks) * FRAG_HALVES + lane * 16);
                    acc = __builtin_amdgcn_wmma_f32_16x16x32_f16(
                            false, a, false, b, (short)0, acc, false, false);
                }
                const float bv = qkv_b[gnt * 16 + cl];
                const int colOff = (wv == 2)
                    ? ((nt >> 1) * HD + (nt & 1) * 16 + cl) * PP   // Vt: [hd-row][token]
                    : nt * 16 + cl;                                // Q/K: row-major
                #pragma unroll
                for (int r = 0; r < 8; ++r) {
                    const int row = mt * 16 + r + rsel;            // token index
                    dstBase[row * rowMul + colOff] = (_Float16)(acc[r] + bv);
                }
            }
        }
    }
    __syncthreads();

    // ---- stage 3: S = Q K^T, masked softmax, store P (wave = head) ----
    const int h = wv;
    const float scale = 0.17677669529663687f;        // 1/sqrt(32)
    _Float16* Ph = sP + h * LP * PP;
    for (int mt = 0; mt < 4; ++mt) {
        v16h aq = load_a_frag(sQ, XP, mt * 16 + cl, h * HD, ksA);
        v8f s0 = {}, s1 = {}, s2 = {}, s3 = {};
        {
            v16h b;
            b  = load_b_frag(sK + (0 * 16 + cl) * XP + h * HD + ksB);
            s0 = __builtin_amdgcn_wmma_f32_16x16x32_f16(false, aq, false, b, (short)0, s0, false, false);
            b  = load_b_frag(sK + (1 * 16 + cl) * XP + h * HD + ksB);
            s1 = __builtin_amdgcn_wmma_f32_16x16x32_f16(false, aq, false, b, (short)0, s1, false, false);
            b  = load_b_frag(sK + (2 * 16 + cl) * XP + h * HD + ksB);
            s2 = __builtin_amdgcn_wmma_f32_16x16x32_f16(false, aq, false, b, (short)0, s2, false, false);
            b  = load_b_frag(sK + (3 * 16 + cl) * XP + h * HD + ksB);
            s3 = __builtin_amdgcn_wmma_f32_16x16x32_f16(false, aq, false, b, (short)0, s3, false, false);
        }
        const bool v0 = (0 * 16 + cl) < L, v1 = (1 * 16 + cl) < L;
        const bool v2 = (2 * 16 + cl) < L, v3 = (3 * 16 + cl) < L;
        #pragma unroll
        for (int r = 0; r < 8; ++r) {
            float e0 = v0 ? s0[r] * scale : -1e30f;
            float e1 = v1 ? s1[r] * scale : -1e30f;
            float e2 = v2 ? s2[r] * scale : -1e30f;
            float e3 = v3 ? s3[r] * scale : -1e30f;
            float m = fmaxf(fmaxf(e0, e1), fmaxf(e2, e3));
            m = fmaxf(m, __shfl_xor(m, 1));
            m = fmaxf(m, __shfl_xor(m, 2));
            m = fmaxf(m, __shfl_xor(m, 4));
            m = fmaxf(m, __shfl_xor(m, 8));          // reduce across 16-lane half = full row
            e0 = __expf(e0 - m); e1 = __expf(e1 - m);
            e2 = __expf(e2 - m); e3 = __expf(e3 - m);
            float sm = e0 + e1 + e2 + e3;
            sm += __shfl_xor(sm, 1);
            sm += __shfl_xor(sm, 2);
            sm += __shfl_xor(sm, 4);
            sm += __shfl_xor(sm, 8);
            const float inv = __builtin_amdgcn_rcpf(sm);
            const int row = mt * 16 + r + rsel;
            Ph[row * PP + 0 * 16 + cl] = (_Float16)(e0 * inv);
            Ph[row * PP + 1 * 16 + cl] = (_Float16)(e1 * inv);
            Ph[row * PP + 2 * 16 + cl] = (_Float16)(e2 * inv);
            Ph[row * PP + 3 * 16 + cl] = (_Float16)(e3 * inv);
        }
    }
    // no barrier needed: P consumed by the same wave that produced it;
    // Vt was fenced by the stage-2 barrier.

    // ---- stage 4: O_h = P V  (K-dim 64 = 2 steps) ----
    for (int mt = 0; mt < 4; ++mt) {
        #pragma unroll
        for (int nt = 0; nt < 2; ++nt) {
            v8f acc = {};
            #pragma unroll
            for (int ks = 0; ks < 2; ++ks) {
                v16h a = load_a_frag(Ph, PP, mt * 16 + cl, ks * 32, ksA);
                v16h b = load_b_frag(sVt + (h * HD + nt * 16 + cl) * PP + ks * 32 + ksB);
                acc = __builtin_amdgcn_wmma_f32_16x16x32_f16(
                        false, a, false, b, (short)0, acc, false, false);
            }
            #pragma unroll
            for (int r = 0; r < 8; ++r) {
                const int row = mt * 16 + r + rsel;
                sO[row * XP + h * HD + nt * 16 + cl] = (_Float16)acc[r];
            }
        }
    }
    __syncthreads();

    // ---- stage 5: out = O @ proj_w + proj_b, scatter back to image ----
    for (int mt = 0; mt < 4; ++mt) {
        #pragma unroll
        for (int nti = 0; nti < 2; ++nti) {
            const int nt = wv * 2 + nti;             // N-tile in [0,6)
            v8f acc = {};
            #pragma unroll
            for (int ks = 0; ks < 3; ++ks) {
                v16h a = load_a_frag(sO, XP, mt * 16 + cl, ks * 32, ksA);
                v16h b = load_b_frag(wfrag + (size_t)(QKV_FRAGS + nt * 3 + ks) * FRAG_HALVES + lane * 16);
                acc = __builtin_amdgcn_wmma_f32_16x16x32_f16(
                        false, a, false, b, (short)0, acc, false, false);
            }
            const float pb = proj_b[nt * 16 + cl];
            const int  cc  = nt * 16 + cl;
            #pragma unroll
            for (int r = 0; r < 8; ++r) {
                const int t = mt * 16 + r + rsel;
                if (t < L) {
                    const int gy = oy + t / WSZ, gx = ox + t % WSZ;
                    out[xbase + ((size_t)gy * IMG + gx) * C + cc] = acc[r] + pb;
                }
            }
        }
    }
}

// =============================== launcher ===============================
extern "C" void kernel_launch(void* const* d_in, const int* in_sizes, int n_in,
                              void* d_out, int out_size, void* d_ws, size_t ws_size,
                              hipStream_t stream) {
    (void)in_sizes; (void)n_in; (void)out_size; (void)ws_size;
    const float* x      = (const float*)d_in[0];
    const float* qkv_w  = (const float*)d_in[1];
    const float* qkv_b  = (const float*)d_in[2];
    const float* proj_w = (const float*)d_in[3];
    const float* proj_b = (const float*)d_in[4];
    float*       out    = (float*)d_out;
    _Float16*    wfrag  = (_Float16*)d_ws;   // 72 fragments * 1 KB = 73728 B

    pack_weights_kernel<<<QKV_FRAGS + PRJ_FRAGS, 32, 0, stream>>>(qkv_w, proj_w, wfrag);
    win_attn_kernel<<<NWIN, 96, 0, stream>>>(x, wfrag, qkv_b, proj_b, out);
}
